// BasicLSTM_89721866813941
// MI455X (gfx1250) — matrix operand order repaired and checked by reading, not compiled, not run on hardware
//
#include <hip/hip_runtime.h>
#include <math.h>

// ---------------------------------------------------------------------------
// BasicLSTM on MI455X (gfx1250): bf16 WMMA GEMMs + persistent recurrent kernel
//   T=512, B=32, D=1024, H=1024
// Phase 1 (x @ Wx): HBM-store-bound big GEMM, bf16 WMMA, f32 accumulate.
// Phase 2 (recurrence): latency-bound; persistent kernel (8 blocks / 64 waves),
//   Wh (8 MB bf16) resident in 192 MB L2, h staged into LDS per step via the
//   Tensor Data Mover (tensor_load_to_lds + s_wait_tensorcnt).
// ---------------------------------------------------------------------------

typedef __attribute__((ext_vector_type(16))) __bf16       v16bf;
typedef __attribute__((ext_vector_type(8)))  float        v8f;
typedef __attribute__((ext_vector_type(4)))  unsigned int v4u;
typedef __attribute__((ext_vector_type(8)))  int          v8i;
typedef __attribute__((ext_vector_type(4)))  int          v4i;

#define LSTM_T 512
#define LSTM_B 32
#define LSTM_D 1024
#define LSTM_H 1024
#define G4H    4096          // 4 gates * H
#define KCH    32            // 1024 / 32 (K chunks of 32 for bf16 WMMA)

// ------------------------- packed fragment layouts --------------------------
// A fragment (16x32 bf16, M x K), per ISA 7.12.2:
//   lane L (half = L>>4, row M = L&15), elems e:
//     e<8  : K = half*8 + e          (K 0..15)
//     e>=8 : K = 16 + half*8 + (e-8) (K 16..31)
// B fragment (32x16 bf16, K x N):
//   lane L: col N = L&15, half = L>>4; elem e: K = a_kmap(L, e), same order as A
// Packed arrays: [tile][k_chunk][lane][e] -> one contiguous 32B load per lane.

__device__ __forceinline__ int a_kmap(int lane, int e) {
    int half = lane >> 4;
    return (e < 8) ? (half * 8 + e) : (16 + half * 8 + (e - 8));
}

// --------------------------- packing kernels --------------------------------

__global__ void pack_weights(const float* __restrict__ Wii, const float* __restrict__ Whi,
                             const float* __restrict__ Wif, const float* __restrict__ Whf,
                             const float* __restrict__ Wig, const float* __restrict__ Whg,
                             const float* __restrict__ Wio, const float* __restrict__ Who,
                             __bf16* __restrict__ WxP, __bf16* __restrict__ WhP) {
    size_t idx = (size_t)blockIdx.x * 256 + threadIdx.x;          // 0 .. 2*4M
    const size_t per = (size_t)G4H * LSTM_D;                       // 4,194,304
    bool isH   = idx >= per;
    size_t l   = isH ? (idx - per) : idx;
    int n_tile = (int)(l / 16384);                                 // 0..255
    int r      = (int)(l % 16384);
    int kc     = r / 512;                                          // 0..31
    int lane   = (r % 512) / 16;
    int e      = r % 16;
    int n      = n_tile * 16 + (lane & 15);                        // 0..4095
    int k      = kc * 32 + a_kmap(lane, e);                        // B-fragment K map
    int gate   = n >> 10;
    int nc     = n & 1023;
    const float* Wx[4] = {Wii, Wif, Wig, Wio};
    const float* Wh[4] = {Whi, Whf, Whg, Who};
    const float* src = isH ? Wh[gate] : Wx[gate];
    float v = src[(size_t)k * LSTM_H + nc];
    (isH ? WhP : WxP)[l] = (__bf16)v;
}

__global__ void pack_bias(const float* __restrict__ bi, const float* __restrict__ bf,
                          const float* __restrict__ bg, const float* __restrict__ bo,
                          float* __restrict__ bias) {
    int i = blockIdx.x * 256 + threadIdx.x;                        // 0..4095
    const float* bs[4] = {bi, bf, bg, bo};
    bias[i] = bs[i >> 10][i & 1023];
}

__global__ void pack_x(const float* __restrict__ x, __bf16* __restrict__ xP) {
    size_t idx = (size_t)blockIdx.x * 256 + threadIdx.x;           // 0..16,777,215
    int m_tile = (int)(idx / 16384);                               // 0..1023
    int r      = (int)(idx % 16384);
    int kc     = r / 512;
    int lane   = (r % 512) / 16;
    int e      = r % 16;
    int m      = m_tile * 16 + (lane & 15);
    int k      = kc * 32 + a_kmap(lane, e);
    xP[idx] = (__bf16)x[(size_t)m * LSTM_D + k];
}

__global__ void init_state(const float* __restrict__ h0, const float* __restrict__ c0,
                           __bf16* __restrict__ hP, float* __restrict__ cbuf,
                           unsigned* __restrict__ bar) {
    int idx = blockIdx.x * 256 + threadIdx.x;                      // 0..32767
    cbuf[idx] = c0[idx];
    int m_tile = idx / 16384;                                      // 0..1
    int r      = idx % 16384;
    int kc     = r / 512;
    int lane   = (r % 512) / 16;
    int e      = r % 16;
    int m      = m_tile * 16 + (lane & 15);
    int k      = kc * 32 + a_kmap(lane, e);
    hP[idx] = (__bf16)h0[m * LSTM_H + k];
    if (blockIdx.x == 0 && threadIdx.x < 64) bar[threadIdx.x] = 0u;
}

// ------------------------ phase 1: x projections ----------------------------
// xg[m][n] = bias[n] + sum_k x[m][k] * Wx[k][n]   (M=16384, N=4096, K=1024)
// Block = 8 waves, tile 128(M) x 128(N); wave tile 32x64 (2x4 WMMA tiles).

__global__ void __launch_bounds__(256)
gemm_x(const __bf16* __restrict__ xP, const __bf16* __restrict__ WxP,
       const float* __restrict__ bias, float* __restrict__ xg) {
    const int lane = threadIdx.x & 31;
    const int w    = threadIdx.x >> 5;
    const int wm   = w & 3;
    const int wn   = w >> 2;
    const int mt0  = blockIdx.y * 8 + wm * 2;
    const int nt0  = blockIdx.x * 8 + wn * 4;

    v8f acc[2][4];
    for (int i = 0; i < 2; ++i)
        for (int j = 0; j < 4; ++j)
            acc[i][j] = {};

    for (int kc = 0; kc < KCH; ++kc) {
        v16bf a[2], b[4];
        for (int i = 0; i < 2; ++i)
            a[i] = *(const v16bf*)(xP + ((((size_t)(mt0 + i)) * KCH + kc) * 32 + lane) * 16);
        for (int j = 0; j < 4; ++j)
            b[j] = *(const v16bf*)(WxP + ((((size_t)(nt0 + j)) * KCH + kc) * 32 + lane) * 16);
        for (int i = 0; i < 2; ++i)
            for (int j = 0; j < 4; ++j)
                acc[i][j] = __builtin_amdgcn_wmma_f32_16x16x32_bf16(
                    false, a[i], false, b[j], (short)0, acc[i][j], false, false);
    }

    const int col  = lane & 15;
    const int rsel = (lane >> 4) * 8;
    for (int i = 0; i < 2; ++i)
        for (int j = 0; j < 4; ++j) {
            int n     = (nt0 + j) * 16 + col;
            float bn  = bias[n];
            int mbase = (mt0 + i) * 16 + rsel;
            for (int v = 0; v < 8; ++v)
                xg[(size_t)(mbase + v) * G4H + n] = acc[i][j][v] + bn;
        }
}

// ----------------------- device-wide split barrier --------------------------

__device__ __forceinline__ void grid_barrier(unsigned nblocks,
                                             unsigned* cnt, unsigned* gen) {
    __threadfence();
    __syncthreads();
    if (threadIdx.x == 0) {
        unsigned g = __hip_atomic_load(gen, __ATOMIC_RELAXED, __HIP_MEMORY_SCOPE_AGENT);
        unsigned arrived = __hip_atomic_fetch_add(cnt, 1u, __ATOMIC_ACQ_REL,
                                                  __HIP_MEMORY_SCOPE_AGENT);
        if (arrived == nblocks - 1u) {
            __hip_atomic_store(cnt, 0u, __ATOMIC_RELAXED, __HIP_MEMORY_SCOPE_AGENT);
            __hip_atomic_fetch_add(gen, 1u, __ATOMIC_RELEASE, __HIP_MEMORY_SCOPE_AGENT);
        } else {
            while (__hip_atomic_load(gen, __ATOMIC_ACQUIRE,
                                     __HIP_MEMORY_SCOPE_AGENT) == g) {
                __builtin_amdgcn_s_sleep(8);
            }
        }
    }
    __syncthreads();
    __threadfence();
}

// ------------------- TDM: DMA packed h (64 KB) into LDS ---------------------
// D# per CDNA5 ISA: flat 1-D tile, data_size=2B,
// tile_dim0 = tensor_dim0 = 32768 elems, tile/tensor_dim1 = 1, type=2,
// count=1 (valid user descriptor), no pad/iterate/multicast.

__device__ __forceinline__ void tdm_load_h_to_lds(const __bf16* hPg, unsigned ldsOff) {
    unsigned long long ga = (unsigned long long)(uintptr_t)hPg;
    v4u g0;
    g0.x = 1u;                                                 // count=1
    g0.y = ldsOff;                                             // lds_addr (bytes)
    g0.z = (unsigned)(ga & 0xFFFFFFFFu);                       // global_addr[31:0]
    g0.w = (unsigned)((ga >> 32) & 0x01FFFFFFu) | 0x80000000u; // ga[56:32] | type=2
    v8i g1;
    g1[0] = 0x00010000;            // data_size=1 (2 bytes); workgroup_mask=0
    g1[1] = (int)0x80000000u;      // tensor_dim0[15:0]=32768 in bits [31:16]
    g1[2] = 0x00010000;            // tensor_dim0[31:16]=0; tensor_dim1[15:0]=1
    g1[3] = (int)0x80000000u;      // tensor_dim1[31:16]=0; tile_dim0=32768
    g1[4] = 0x00000001;            // tile_dim1=1; tile_dim2=0
    g1[5] = 32768;                 // tensor_dim0_stride[31:0]
    g1[6] = 0;                     // stride high bits / dim1_stride low
    g1[7] = 0;
    v4i z4 = {0, 0, 0, 0};
#if defined(__clang_major__) && (__clang_major__ >= 23)
    v8i z8 = {0, 0, 0, 0, 0, 0, 0, 0};
    __builtin_amdgcn_tensor_load_to_lds(g0, g1, z4, z4, z8, 0);
#else
    __builtin_amdgcn_tensor_load_to_lds(g0, g1, z4, z4, 0);
#endif
}

// ------------------- phase 2: persistent recurrent kernel -------------------
// 8 blocks x 256 threads = 64 waves. Wave w owns 64 gate columns (4 WMMA
// n-tiles) and both M-tiles (full B=32). Per step:
//   TDM h->LDS, s_wait_tensorcnt, block barrier,
//   WMMA GEMM (A from LDS via ds_load, B from L2-resident WhP) -> gbuf,
//   grid barrier, fused gates + c/h update + bf16 repack of h, grid barrier.

__global__ void __launch_bounds__(256)
lstm_recurrent(const __bf16* __restrict__ WhP, const float* __restrict__ xg,
               __bf16* hP, float* cbuf, float* gbuf, unsigned* bar,
               float* __restrict__ out) {
    __shared__ __align__(64) __bf16 hLds[LSTM_B * LSTM_H];     // 64 KB packed h
    const int lane = threadIdx.x & 31;
    const int w    = threadIdx.x >> 5;
    const int wg   = blockIdx.x * 8 + w;     // 0..63
    const int nt0  = wg * 4;
    unsigned* cnt  = bar;
    unsigned* gen  = bar + 16;
    const unsigned ldsOff = (unsigned)(uintptr_t)&hLds[0];

    for (int t = 0; t < LSTM_T; ++t) {
        // ---- stage packed h into LDS via Tensor Data Mover (wave 0) ----
        if (w == 0) {
            tdm_load_h_to_lds(hP, ldsOff);
            __builtin_amdgcn_s_wait_tensorcnt(0);
        }
        __syncthreads();

        // ---- GEMM: gbuf[32][4096] = h @ Wh (pre-activations, no bias) ----
        v8f acc[2][4];
        for (int i = 0; i < 2; ++i)
            for (int j = 0; j < 4; ++j)
                acc[i][j] = {};

        for (int kc = 0; kc < KCH; ++kc) {
            v16bf a[2], b[4];
            a[0] = *(const v16bf*)&hLds[((size_t)(0 * KCH + kc) * 32 + lane) * 16];
            a[1] = *(const v16bf*)&hLds[((size_t)(1 * KCH + kc) * 32 + lane) * 16];
            for (int j = 0; j < 4; ++j)
                b[j] = *(const v16bf*)(WhP + ((((size_t)(nt0 + j)) * KCH + kc) * 32 + lane) * 16);
            for (int i = 0; i < 2; ++i)
                for (int j = 0; j < 4; ++j)
                    acc[i][j] = __builtin_amdgcn_wmma_f32_16x16x32_bf16(
                        false, a[i], false, b[j], (short)0, acc[i][j], false, false);
        }

        const int col  = lane & 15;
        const int rsel = (lane >> 4) * 8;
        for (int i = 0; i < 2; ++i)
            for (int j = 0; j < 4; ++j) {
                int n     = (nt0 + j) * 16 + col;
                int mbase = i * 16 + rsel;
                for (int v = 0; v < 8; ++v)
                    gbuf[(size_t)(mbase + v) * G4H + n] = acc[i][j][v];
            }

        grid_barrier(8u, cnt, gen);

        // ---- fused gates + state update + output + h repack ----
        const int tg = blockIdx.x * 256 + threadIdx.x;             // 0..2047
        for (int r = 0; r < 16; ++r) {
            int idx = tg * 16 + r;                                 // 0..32767
            int bb  = idx >> 10;
            int j   = idx & 1023;
            const float* xrow = xg + ((size_t)t * LSTM_B + bb) * G4H;
            const float* grow = gbuf + (size_t)bb * G4H;
            float pi = xrow[j]        + grow[j];
            float pf = xrow[j + 1024] + grow[j + 1024];
            float pg = xrow[j + 2048] + grow[j + 2048];
            float po = xrow[j + 3072] + grow[j + 3072];
            float it = 1.0f / (1.0f + __expf(-pi));
            float ft = 1.0f / (1.0f + __expf(-pf));
            float gt = tanhf(pg);
            float ot = 1.0f / (1.0f + __expf(-po));
            float cn = ft * cbuf[idx] + it * gt;
            float hn = ot * tanhf(cn);
            cbuf[idx] = cn;
            // hidden_seq is [B, T, H]
            out[(size_t)bb * (LSTM_T * LSTM_H) + (size_t)t * LSTM_H + j] = hn;
            // repack h into bf16 A-fragment layout for next step
            int m_tile = bb >> 4;
            int kc2    = j >> 5;
            int kr     = j & 31;
            int half, e;
            if (kr < 16) { half = kr >> 3; e = kr & 7; }
            else         { int k2 = kr - 16; half = k2 >> 3; e = 8 + (k2 & 7); }
            int ln = half * 16 + (bb & 15);
            hP[((((size_t)m_tile) * KCH + kc2) * 32 + ln) * 16 + e] = (__bf16)hn;
            if (t == LSTM_T - 1) {
                out[(size_t)LSTM_B * LSTM_T * LSTM_H + idx]         = hn;  // h_T
                out[(size_t)LSTM_B * LSTM_T * LSTM_H + 32768 + idx] = cn;  // c_T
            }
        }

        grid_barrier(8u, cnt, gen);
    }
}

// --------------------------------- launch -----------------------------------

extern "C" void kernel_launch(void* const* d_in, const int* in_sizes, int n_in,
                              void* d_out, int out_size, void* d_ws, size_t ws_size,
                              hipStream_t stream) {
    const float* x   = (const float*)d_in[0];
    const float* h0  = (const float*)d_in[1];
    const float* c0  = (const float*)d_in[2];
    const float* Wii = (const float*)d_in[3];
    const float* Whi = (const float*)d_in[4];
    const float* bi  = (const float*)d_in[5];
    const float* Wif = (const float*)d_in[6];
    const float* Whf = (const float*)d_in[7];
    const float* bf  = (const float*)d_in[8];
    const float* Wig = (const float*)d_in[9];
    const float* Whg = (const float*)d_in[10];
    const float* bg  = (const float*)d_in[11];
    const float* Wio = (const float*)d_in[12];
    const float* Who = (const float*)d_in[13];
    const float* bo  = (const float*)d_in[14];
    float* out = (float*)d_out;

    char* ws = (char*)d_ws;
    __bf16*   xP   = (__bf16*)(ws);                 // 32 MB  : packed x
    __bf16*   WxP  = (__bf16*)(ws + 33554432);      //  8 MB  : packed Wx (4 gates)
    __bf16*   WhP  = (__bf16*)(ws + 41943040);      //  8 MB  : packed Wh (4 gates)
    float*    bias = (float*) (ws + 50331648);      // 16 KB
    __bf16*   hP   = (__bf16*)(ws + 50348032);      // 64 KB  : packed h (bf16)
    float*    cbuf = (float*) (ws + 50413568);      // 128 KB : c state
    float*    gbuf = (float*) (ws + 50544640);      // 512 KB : h@Wh pre-acts
    unsigned* bar  = (unsigned*)(ws + 51068928);    // 256 B  : grid barrier
    float*    xg   = (float*) (ws + 51069184);      // 256 MB : x projections

    pack_weights<<<32768, 256, 0, stream>>>(Wii, Whi, Wif, Whf, Wig, Whg, Wio, Who, WxP, WhP);
    pack_bias   <<<16,    256, 0, stream>>>(bi, bf, bg, bo, bias);
    pack_x      <<<65536, 256, 0, stream>>>(x, xP);
    init_state  <<<128,   256, 0, stream>>>(h0, c0, hP, cbuf, bar);
    gemm_x      <<<dim3(32, 128), 256, 0, stream>>>(xP, WxP, bias, xg);
    lstm_recurrent<<<8,   256, 0, stream>>>(WhP, xg, hP, cbuf, gbuf, bar, out);
}
